// ProfessionalNetworkGNN_67929202753824
// MI455X (gfx1250) — compile-verified, matrix operand order checked by reading, not run on hardware
//
#include <hip/hip_runtime.h>
#include <hip/hip_bf16.h>
#include <hip/hip_fp16.h>
#include <math.h>

typedef __attribute__((ext_vector_type(16))) _Float16 v16h;
typedef __attribute__((ext_vector_type(8)))  float    v8f;

#define CDIV(a, b) (((a) + (b) - 1) / (b))

// ---------------- helpers ----------------

__device__ __forceinline__ float lrelu(float x) { return x > 0.f ? x : 0.2f * x; }

// order-preserving float <-> uint encoding for atomicMax on floats
__device__ __forceinline__ unsigned enc_f(float f) {
  unsigned u = __float_as_uint(f);
  return (u & 0x80000000u) ? ~u : (u | 0x80000000u);
}
__device__ __forceinline__ float dec_f(unsigned u) {
  return (u & 0x80000000u) ? __uint_as_float(u & 0x7fffffffu) : __uint_as_float(~u);
}

// K index for 16-bit A/B WMMA fragments: VGPR pair v (0..7), lane-half (0/1)
__device__ __forceinline__ int wmma_k(int v, int half) {
  return ((v & 3) << 1) + ((v >> 2) << 4) + (half << 3);
}

__device__ __forceinline__ float4 relu4(float4 p) {
  p.x = fmaxf(p.x, 0.f); p.y = fmaxf(p.y, 0.f);
  p.z = fmaxf(p.z, 0.f); p.w = fmaxf(p.w, 0.f);
  return p;
}

// pack two contiguous float4 chunks into 8 f16 lanes of a fragment
__device__ __forceinline__ void cvt8(v16h& a, int o, float4 u, float4 v) {
  a[o + 0] = (_Float16)u.x; a[o + 1] = (_Float16)u.y;
  a[o + 2] = (_Float16)u.z; a[o + 3] = (_Float16)u.w;
  a[o + 4] = (_Float16)v.x; a[o + 5] = (_Float16)v.y;
  a[o + 6] = (_Float16)v.z; a[o + 7] = (_Float16)v.w;
}

// ---------------- weight pre-pack: W[K,Nout] f32 -> B fragments f16 ----------------
// fragment fid = kt*(Nout/16)+ct holds a 32(K) x 16(N) tile, 512 f16, lane-major:
// lane L stores 16 f16 at out[fid*512 + L*16]: pairs (W[k][n], W[k+1][n]) per VGPR slot.
__global__ void k_pack_w(const float* __restrict__ W, _Float16* __restrict__ out,
                         int K, int Nout) {
  int idx = blockIdx.x * blockDim.x + threadIdx.x;
  int lane = idx & 31;
  int fid = idx >> 5;
  int nct = Nout >> 4;
  int nkt = K >> 5;
  if (fid >= nct * nkt) return;
  int ct = fid % nct, kt = fid / nct;
  int half = lane >> 4;
  int n = ct * 16 + (lane & 15);
  _Float16* op = out + (size_t)fid * 512 + lane * 16;
#pragma unroll
  for (int v = 0; v < 8; ++v) {
    int k = kt * 32 + wmma_k(v, half);
    op[2 * v]     = (_Float16)W[(size_t)k * Nout + n];
    op[2 * v + 1] = (_Float16)W[(size_t)(k + 1) * Nout + n];
  }
}

// ---------------- WMMA GEMM: C[M,Nout] = act(A[M,K]) @ W[K,Nout] ----------------
// Wp = pre-packed f16 B fragments. grid.x = M/16 row tiles, (grid.y*4 + wave) = col tile.
template <int K, bool RELU>
__global__ __launch_bounds__(128) void k_gemm_wmma(const float* __restrict__ A,
                                                   const _Float16* __restrict__ Wp,
                                                   float* __restrict__ C, int Nout) {
  const int lane = threadIdx.x & 31;
  const int wave = threadIdx.x >> 5;
  const int m    = lane & 15;
  const int half = lane >> 4;
  const int row0 = blockIdx.x << 4;
  const int nct  = Nout >> 4;
  const int ct   = (blockIdx.y << 2) + wave;
  if (ct >= nct) return;  // wave-uniform

  v8f c = {};
  const float*    arow = A + (size_t)(row0 + m) * K;
  const _Float16* bp   = Wp + (size_t)ct * 512 + lane * 16;
#pragma unroll
  for (int kt = 0; kt < (K >> 5); ++kt) {
    const int base = (kt << 5) + (half << 3);
    float4 p0 = *(const float4*)(arow + base);
    float4 p1 = *(const float4*)(arow + base + 4);
    float4 q0 = *(const float4*)(arow + base + 16);
    float4 q1 = *(const float4*)(arow + base + 20);
    if (RELU) { p0 = relu4(p0); p1 = relu4(p1); q0 = relu4(q0); q1 = relu4(q1); }
    v16h a;
    cvt8(a, 0, p0, p1);
    cvt8(a, 8, q0, q1);
    const v16h b = *(const v16h*)(bp + (size_t)kt * nct * 512);
    c = __builtin_amdgcn_wmma_f32_16x16x32_f16(false, a, false, b, (short)0, c,
                                               false, false);
  }
#pragma unroll
  for (int r = 0; r < 8; ++r) {
    const int row = row0 + r + (half << 3);
    C[(size_t)row * Nout + (ct << 4) + m] = c[r];
  }
}

// ---------------- elementwise / graph kernels ----------------

__global__ void k_fill(float* __restrict__ p, float v, size_t n) {
  size_t i = (size_t)blockIdx.x * blockDim.x + threadIdx.x;
  if (i < n) p[i] = v;
}

__global__ void k_deg(const int* __restrict__ dst, float* __restrict__ deg, int E) {
  int e = blockIdx.x * blockDim.x + threadIdx.x;
  if (e < E) atomicAdd(&deg[dst[e]], 1.0f);
}

__global__ void k_rsqrt_ip(float* __restrict__ d, int n) {
  int i = blockIdx.x * blockDim.x + threadIdx.x;
  if (i < n) d[i] = rsqrtf(d[i]);  // deg >= 1 (self-loop)
}

// out[n,f] = h[n,f]*dinv[n]^2 + b[f]  (self-loop contribution + bias)
__global__ void k_gcn_init(const float* __restrict__ h, const float* __restrict__ b,
                           const float* __restrict__ dinv, float* __restrict__ out,
                           size_t total, int F) {
  size_t i = (size_t)blockIdx.x * blockDim.x + threadIdx.x;
  if (i >= total) return;
  size_t n = i / F;
  int f = (int)(i % F);
  float di = dinv[n];
  out[i] = h[i] * di * di + b[f];
}

// out[dst] += h[src] * dinv[src]*dinv[dst], 4 features per thread
__global__ void k_gcn_scatter(const float* __restrict__ h, const float* __restrict__ dinv,
                              const int* __restrict__ src, const int* __restrict__ dst,
                              float* __restrict__ out, int E, int F) {
  const int chunks = F >> 2;
  size_t idx = (size_t)blockIdx.x * blockDim.x + threadIdx.x;
  int e = (int)(idx / chunks);
  if (e >= E) return;
  int c = (int)(idx % chunks);
  int s = src[e], d = dst[e];
  float w = dinv[s] * dinv[d];
  const float* hp = h + (size_t)s * F + c * 4;
  float* op = out + (size_t)d * F + c * 4;
#pragma unroll
  for (int i = 0; i < 4; ++i) atomicAdd(op + i, hp[i] * w);
}

// al_s[n,h] = <hg[n,h,:], a_src[h,:]>, same for al_d
__global__ void k_gat_attn(const float* __restrict__ hg, const float* __restrict__ a_src,
                           const float* __restrict__ a_dst, float* __restrict__ als,
                           float* __restrict__ ald, int N) {
  int idx = blockIdx.x * blockDim.x + threadIdx.x;
  int n = idx >> 2, h = idx & 3;
  if (n >= N) return;
  const float* hp = hg + (size_t)n * 256 + h * 64;
  const float* as = a_src + h * 64;
  const float* ad = a_dst + h * 64;
  float s = 0.f, d = 0.f;
#pragma unroll 8
  for (int f = 0; f < 64; ++f) { s += hp[f] * as[f]; d += hp[f] * ad[f]; }
  als[idx] = s;
  ald[idx] = d;
}

__global__ void k_gat_minit(const float* __restrict__ als, const float* __restrict__ ald,
                            unsigned* __restrict__ menc, int n4) {
  int i = blockIdx.x * blockDim.x + threadIdx.x;
  if (i < n4) menc[i] = enc_f(lrelu(als[i] + ald[i]));  // self edge
}

__global__ void k_gat_mscatter(const float* __restrict__ als, const float* __restrict__ ald,
                               const int* __restrict__ src, const int* __restrict__ dst,
                               unsigned* __restrict__ menc, int E) {
  int idx = blockIdx.x * blockDim.x + threadIdx.x;
  int e = idx >> 2, h = idx & 3;
  if (e >= E) return;
  int s = src[e], d = dst[e];
  atomicMax(&menc[d * 4 + h], enc_f(lrelu(als[s * 4 + h] + ald[d * 4 + h])));
}

// z[n,h] = exp(e_self - m)   (self-loop contribution)
__global__ void k_gat_zinit(const float* __restrict__ als, const float* __restrict__ ald,
                            const unsigned* __restrict__ menc, float* __restrict__ z,
                            int n4) {
  int i = blockIdx.x * blockDim.x + threadIdx.x;
  if (i < n4) z[i] = expf(lrelu(als[i] + ald[i]) - dec_f(menc[i]));
}

// acc[n,h,f] = exp(e_self - m) * hg[n,h,f]
__global__ void k_gat_accinit(const float* __restrict__ als, const float* __restrict__ ald,
                              const unsigned* __restrict__ menc,
                              const float* __restrict__ hg, float* __restrict__ acc,
                              size_t total) {
  size_t i = (size_t)blockIdx.x * blockDim.x + threadIdx.x;
  if (i >= total) return;
  size_t n = i >> 8;
  int h = (int)((i >> 6) & 3);
  float ex = expf(lrelu(als[n * 4 + h] + ald[n * 4 + h]) - dec_f(menc[n * 4 + h]));
  acc[i] = ex * hg[i];
}

__global__ void k_gat_zscatter(const float* __restrict__ als, const float* __restrict__ ald,
                               const unsigned* __restrict__ menc,
                               const int* __restrict__ src, const int* __restrict__ dst,
                               float* __restrict__ z, int E) {
  int idx = blockIdx.x * blockDim.x + threadIdx.x;
  int e = idx >> 2, h = idx & 3;
  if (e >= E) return;
  int s = src[e], d = dst[e];
  float ex = expf(lrelu(als[s * 4 + h] + ald[d * 4 + h]) - dec_f(menc[d * 4 + h]));
  atomicAdd(&z[d * 4 + h], ex);
}

// acc[dst,h,f..f+3] += ex * hg[src,h,f..f+3]
__global__ void k_gat_accscatter(const float* __restrict__ als, const float* __restrict__ ald,
                                 const unsigned* __restrict__ menc,
                                 const float* __restrict__ hg,
                                 const int* __restrict__ src, const int* __restrict__ dst,
                                 float* __restrict__ acc, int E) {
  size_t idx = (size_t)blockIdx.x * blockDim.x + threadIdx.x;
  int e = (int)(idx >> 6);
  if (e >= E) return;
  int r = (int)(idx & 63);
  int h = r >> 4;
  int f = (r & 15) << 2;
  int s = src[e], d = dst[e];
  float ex = expf(lrelu(als[s * 4 + h] + ald[d * 4 + h]) - dec_f(menc[d * 4 + h]));
  const float* hp = hg + (size_t)s * 256 + h * 64 + f;
  float* ap = acc + (size_t)d * 256 + h * 64 + f;
#pragma unroll
  for (int i = 0; i < 4; ++i) atomicAdd(ap + i, ex * hp[i]);
}

// emb[n,f] = mean_h( acc[n,h,f] / z[n,h] ) + bg[f]
__global__ void k_gat_final(const float* __restrict__ acc, const float* __restrict__ z,
                            const float* __restrict__ bg, float* __restrict__ emb,
                            size_t total) {
  size_t i = (size_t)blockIdx.x * blockDim.x + threadIdx.x;
  if (i >= total) return;
  size_t n = i >> 6;
  int f = (int)(i & 63);
  float s = 0.f;
#pragma unroll
  for (int h = 0; h < 4; ++h) s += acc[n * 256 + h * 64 + f] / z[n * 4 + h];
  emb[i] = s * 0.25f + bg[f];
}

// ---------------- fused edge-predictor MLP ----------------
// block = 256 threads = 8 waves; each block handles 16 edges.
// Layer 1: wave w -> 16x16 tile of hid = relu([emb_src|emb_dst] @ P1 + pb1) via WMMA
// (K=128 fully unrolled, packed-f16 B), staged in LDS. Layer 2: 16 lanes/edge dot
// with P2, shfl_xor tree reduction, sigmoid.
__global__ __launch_bounds__(256) void k_edge_mlp(const float* __restrict__ emb,
                                                  const int* __restrict__ src,
                                                  const int* __restrict__ dst,
                                                  const _Float16* __restrict__ P1p,
                                                  const float* __restrict__ pb1,
                                                  const float* __restrict__ P2,
                                                  const float* __restrict__ pb2,
                                                  float* __restrict__ preds, int E) {
  __shared__ float hid_s[16][132];  // +4 pad to spread LDS banks
  const int lane = threadIdx.x & 31;
  const int wave = threadIdx.x >> 5;
  const int m    = lane & 15;
  const int half = lane >> 4;
  const int e0   = blockIdx.x << 4;

  int e = e0 + m;
  if (e >= E) e = E - 1;  // clamp (writes are guarded)
  const float* srow = emb + (size_t)src[e] * 64;
  const float* drow = emb + (size_t)dst[e] * 64;

  const _Float16* bp = P1p + (size_t)wave * 512 + lane * 16;  // nct = 8
  v8f c = {};
#pragma unroll
  for (int kt = 0; kt < 4; ++kt) {
    // lane's K-slice = two 8-float chunks; chunks start at multiples of 8 so they
    // never straddle the src|dst concat boundary at 64.
    const int b1 = (kt << 5) + (half << 3);
    const int b2 = b1 + 16;
    const float* c1 = (b1 < 64) ? srow + b1 : drow + (b1 - 64);
    const float* c2 = (b2 < 64) ? srow + b2 : drow + (b2 - 64);
    float4 p0 = *(const float4*)(c1);
    float4 p1 = *(const float4*)(c1 + 4);
    float4 q0 = *(const float4*)(c2);
    float4 q1 = *(const float4*)(c2 + 4);
    v16h a;
    cvt8(a, 0, p0, p1);
    cvt8(a, 8, q0, q1);
    const v16h b = *(const v16h*)(bp + (size_t)kt * 8 * 512);
    c = __builtin_amdgcn_wmma_f32_16x16x32_f16(false, a, false, b, (short)0, c,
                                               false, false);
  }
  const int col = (wave << 4) + m;
#pragma unroll
  for (int r = 0; r < 8; ++r) {
    const int row = r + (half << 3);  // edge-local row
    hid_s[row][col] = fmaxf(c[r] + pb1[col], 0.f);
  }
  __syncthreads();

  const int el = threadIdx.x >> 4;  // edge-local 0..15
  const int g  = threadIdx.x & 15;  // 16 lanes per edge
  float s = 0.f;
#pragma unroll
  for (int i = 0; i < 8; ++i) {
    const int f = g * 8 + i;
    s += hid_s[el][f] * P2[f];
  }
#pragma unroll
  for (int off = 8; off >= 1; off >>= 1) s += __shfl_xor(s, off, 16);
  if (g == 0) {
    const int ee = e0 + el;
    if (ee < E) preds[ee] = 1.f / (1.f + expf(-(s + pb2[0])));
  }
}

// ---------------- launch ----------------

extern "C" void kernel_launch(void* const* d_in, const int* in_sizes, int n_in,
                              void* d_out, int out_size, void* d_ws, size_t ws_size,
                              hipStream_t stream) {
  const int N = 100000, E = 1600000;

  const float* x   = (const float*)d_in[0];
  const int*  eidx = (const int*)d_in[1];
  const int*  src  = eidx;
  const int*  dst  = eidx + E;
  const float* W0 = (const float*)d_in[2];
  const float* b0 = (const float*)d_in[3];
  const float* W1 = (const float*)d_in[4];
  const float* b1 = (const float*)d_in[5];
  const float* W2 = (const float*)d_in[6];
  const float* b2 = (const float*)d_in[7];
  const float* Wg = (const float*)d_in[8];
  const float* a_src = (const float*)d_in[9];
  const float* a_dst = (const float*)d_in[10];
  const float* bg = (const float*)d_in[11];
  const float* P1 = (const float*)d_in[12];
  const float* pb1 = (const float*)d_in[13];
  const float* P2 = (const float*)d_in[14];
  const float* pb2 = (const float*)d_in[15];

  float* emb   = (float*)d_out;                   // [N,64]
  float* preds = (float*)d_out + (size_t)N * 64;  // [E]

  // workspace layout (floats)
  float* ws   = (float*)d_ws;
  float* dinv = ws;                         // N
  float* bufA = dinv + N;                   // N*128
  float* bufB = bufA + (size_t)N * 128;     // N*128
  float* hg   = bufB + (size_t)N * 128;     // N*256
  float* als  = hg + (size_t)N * 256;       // N*4
  float* ald  = als + (size_t)N * 4;        // N*4
  unsigned* menc = (unsigned*)(ald + (size_t)N * 4);  // N*4
  float* z    = (float*)(menc + (size_t)N * 4);       // N*4
  float* acc  = z + (size_t)N * 4;          // N*256
  // packed f16 weights (B fragments), after acc (offset is 32B-aligned)
  _Float16* W0p = (_Float16*)(acc + (size_t)N * 256);  // 8ct*4kt*512 = 16384
  _Float16* W1p = W0p + 16384;                         // 16384
  _Float16* W2p = W1p + 16384;                         // 4ct*4kt*512 = 8192
  _Float16* Wgp = W2p + 8192;                          // 16ct*2kt*512 = 16384
  _Float16* P1p = Wgp + 16384;                         // 8ct*4kt*512 = 16384

  const int T = 256;
  dim3 gB(128);

  // pre-pack weights into WMMA B-fragment order (f16)
  k_pack_w<<<CDIV(8 * 4 * 32, T), T, 0, stream>>>(W0, W0p, 128, 128);
  k_pack_w<<<CDIV(8 * 4 * 32, T), T, 0, stream>>>(W1, W1p, 128, 128);
  k_pack_w<<<CDIV(4 * 4 * 32, T), T, 0, stream>>>(W2, W2p, 128, 64);
  k_pack_w<<<CDIV(16 * 2 * 32, T), T, 0, stream>>>(Wg, Wgp, 64, 256);
  k_pack_w<<<CDIV(8 * 4 * 32, T), T, 0, stream>>>(P1, P1p, 128, 128);

  // degree normalization: deg = 1 (self loop) + indeg; dinv = deg^-1/2
  k_fill<<<CDIV(N, T), T, 0, stream>>>(dinv, 1.0f, (size_t)N);
  k_deg<<<CDIV(E, T), T, 0, stream>>>(dst, dinv, E);
  k_rsqrt_ip<<<CDIV(N, T), T, 0, stream>>>(dinv, N);

  // ---- GCN layer 1: h = x@W0 ; out1 = aggregate + b0 (relu folded into next load)
  k_gemm_wmma<128, false><<<dim3(N / 16, 2), gB, 0, stream>>>(x, W0p, bufA, 128);
  k_gcn_init<<<CDIV((size_t)N * 128, T), T, 0, stream>>>(bufA, b0, dinv, bufB,
                                                         (size_t)N * 128, 128);
  k_gcn_scatter<<<CDIV((size_t)E * 32, T), T, 0, stream>>>(bufA, dinv, src, dst, bufB, E, 128);

  // ---- GCN layer 2 (A = relu(out1))
  k_gemm_wmma<128, true><<<dim3(N / 16, 2), gB, 0, stream>>>(bufB, W1p, bufA, 128);
  k_gcn_init<<<CDIV((size_t)N * 128, T), T, 0, stream>>>(bufA, b1, dinv, bufB,
                                                         (size_t)N * 128, 128);
  k_gcn_scatter<<<CDIV((size_t)E * 32, T), T, 0, stream>>>(bufA, dinv, src, dst, bufB, E, 128);

  // ---- GCN layer 3 (A = relu(out2)), F_out = 64, no relu on output
  k_gemm_wmma<128, true><<<dim3(N / 16, 1), gB, 0, stream>>>(bufB, W2p, bufA, 64);
  k_gcn_init<<<CDIV((size_t)N * 64, T), T, 0, stream>>>(bufA, b2, dinv, bufB,
                                                        (size_t)N * 64, 64);
  k_gcn_scatter<<<CDIV((size_t)E * 16, T), T, 0, stream>>>(bufA, dinv, src, dst, bufB, E, 64);

  // ---- GAT: hg = h3 @ Wg  [N,256]
  k_gemm_wmma<64, false><<<dim3(N / 16, 4), gB, 0, stream>>>(bufB, Wgp, hg, 256);
  k_gat_attn<<<CDIV(N * 4, T), T, 0, stream>>>(hg, a_src, a_dst, als, ald, N);

  // segment softmax over incoming edges (+ self loops) per destination
  k_gat_minit<<<CDIV(N * 4, T), T, 0, stream>>>(als, ald, menc, N * 4);
  k_gat_mscatter<<<CDIV(E * 4, T), T, 0, stream>>>(als, ald, src, dst, menc, E);
  k_gat_zinit<<<CDIV(N * 4, T), T, 0, stream>>>(als, ald, menc, z, N * 4);
  k_gat_accinit<<<CDIV((size_t)N * 256, T), T, 0, stream>>>(als, ald, menc, hg, acc,
                                                            (size_t)N * 256);
  k_gat_zscatter<<<CDIV(E * 4, T), T, 0, stream>>>(als, ald, menc, src, dst, z, E);
  k_gat_accscatter<<<CDIV((size_t)E * 64, T), T, 0, stream>>>(als, ald, menc, hg, src,
                                                              dst, acc, E);
  k_gat_final<<<CDIV((size_t)N * 64, T), T, 0, stream>>>(acc, z, bg, emb,
                                                         (size_t)N * 64);

  // ---- fused edge predictor MLP (dominant GEMM, 52 GFLOP)
  k_edge_mlp<<<CDIV(E, 16), 256, 0, stream>>>(emb, src, dst, P1p, pb1, P2, pb2, preds, E);
}
